// MultiheadLinearEnsembleAttn_86088324481451
// MI455X (gfx1250) — matrix operand order repaired and checked by reading, not compiled, not run on hardware
//
#include <hip/hip_runtime.h>
#include <math.h>

// ---------------------------------------------------------------------------
// MultiheadLinearEnsembleAttn for MI455X (gfx1250, wave32, WMMA + async LDS)
// B=1, T=2048, E=1536, H=16, M=3, HD=32, KVH=8, NREP=2
// ---------------------------------------------------------------------------

typedef __attribute__((ext_vector_type(16))) __bf16 bf16x16;
typedef __attribute__((ext_vector_type(8)))  float  f32x8;
typedef unsigned short u16;

#define T_LEN 2048
#define E_DIM 1536
#define NHEAD 16
#define M_ENS 3
#define HD 32
#define KVH 8

union BfPack { int4 q[2]; bf16x16 v; };

// round-to-nearest-even f32 -> bf16 bits
__device__ __forceinline__ u16 f2bf(float x) {
  unsigned u = __float_as_uint(x);
  unsigned r = u + 0x7FFFu + ((u >> 16) & 1u);
  return (u16)(r >> 16);
}

// A operand: 16x32 bf16, rows along lanes, 16 contiguous K per lane
__device__ __forceinline__ bf16x16 load_a(const u16* base, int ld, int kBase, int lane) {
  int half = lane >> 4, r = lane & 15;
  const u16* p = base + (size_t)r * ld + kBase + half * 16;
  BfPack u;
  u.q[0] = *(const int4*)p;
  u.q[1] = *(const int4*)(p + 8);
  return u.v;
}

// B operand: 32x16 bf16, one output column per lane, 16 contiguous K per lane
__device__ __forceinline__ bf16x16 load_b(const u16* base, int ld, int kBase, int lane) {
  int half = lane >> 4, c = lane & 15;
  const u16* p = base + (size_t)c * ld + kBase + half * 16;
  BfPack u;
  u.q[0] = *(const int4*)p;
  u.q[1] = *(const int4*)(p + 8);
  return u.v;
}

// B operand from an LDS panel laid out [64 cols][32 K] bf16
__device__ __forceinline__ bf16x16 load_b_lds(const u16* buf, int lane) {
  int half = lane >> 4, c = lane & 15;
  const u16* p = buf + c * 32 + half * 16;
  BfPack u;
  u.q[0] = *(const int4*)p;
  u.q[1] = *(const int4*)(p + 8);
  return u.v;
}

__device__ __forceinline__ f32x8 mma(bf16x16 a, bf16x16 b, f32x8 c) {
  return __builtin_amdgcn_wmma_f32_16x16x32_bf16(false, a, false, b, (short)0, c, false, false);
}

// Issue one per-lane 16B async copy global -> LDS (ASYNCcnt-tracked).
__device__ __forceinline__ void async_g2l_b128(u16* ldsDst, const u16* gSrc) {
  unsigned lds = (unsigned)(uintptr_t)ldsDst;            // low 32 bits = LDS offset
  unsigned long long g = (unsigned long long)(uintptr_t)gSrc;
  asm volatile("global_load_async_to_lds_b128 %0, %1, off"
               :: "v"(lds), "v"(g) : "memory");
}

// ---------------------------------------------------------------------------
// Block-cooperative GEMM core: 4 waves x (16M x 64N), shared B panel staged
// through LDS with double-buffered async copies.  sB = u16[2][64*32] (8 KB).
// ---------------------------------------------------------------------------
__device__ __forceinline__ void gemm4_lds(const u16* A, int lda,
                                          const u16* B, int ldb, int K,
                                          int tid, u16* sB, f32x8 acc[4]) {
  int lane = tid & 31;

  // stage 64 cols x 32 K panel (4 KB): 2 x b128 per thread (128 threads)
  auto stage = [&](int kk, u16* buf) {
#pragma unroll
    for (int j = 0; j < 2; ++j) {
      int q = tid + j * 128;          // 256 16-byte chunks
      int c = q >> 2, part = q & 3;   // col, 8-element K sub-chunk
      const u16* src = B + (size_t)c * ldb + kk + part * 8;
      async_g2l_b128(buf + c * 32 + part * 8, src);
      __builtin_prefetch(src + 64, 0, 0);   // pull k+64 panel toward GL2
    }
  };

  stage(0, sB);
  for (int kk = 0; kk < K; kk += 32) {
    u16* cur = sB + (((kk >> 5) & 1) ? 64 * 32 : 0);
    u16* nxt = sB + (((kk >> 5) & 1) ? 0 : 64 * 32);
    if (kk + 32 < K) {
      stage(kk + 32, nxt);                        // prefetch next panel
      asm volatile("s_wait_asynccnt 0x2" ::: "memory");  // current panel done
    } else {
      asm volatile("s_wait_asynccnt 0x0" ::: "memory");
    }
    __syncthreads();                              // panel visible to all waves
    bf16x16 a = load_a(A, lda, kk, lane);
#pragma unroll
    for (int n = 0; n < 4; ++n) {
      bf16x16 b = load_b_lds(cur + n * 16 * 32, lane);
      acc[n] = mma(a, b, acc[n]);
    }
    __syncthreads();                              // reads done before reuse
  }
}

// ---------------------------------------------------------------------------
// f32 -> bf16 bulk convert
// ---------------------------------------------------------------------------
__global__ void k_cvt(const float* __restrict__ src, u16* __restrict__ dst, int n) {
  int i = blockIdx.x * 256 + threadIdx.x;
  if (i < n) dst[i] = f2bf(src[i]);
}

// ---------------------------------------------------------------------------
// Q/K projection + interleaved RoPE (+ q scaling), output [m][head][T][HD] bf16
// ---------------------------------------------------------------------------
__global__ __launch_bounds__(128)
void k_proj_rope(const u16* __restrict__ xb, const u16* __restrict__ wb,
                 const float* __restrict__ cosT, const float* __restrict__ sinT,
                 u16* __restrict__ outb, int Nw, int nHeads, float oscale) {
  __shared__ u16 sB[2 * 64 * 32];
  int tid = threadIdx.x;
  int lane = tid & 31, wv = tid >> 5;
  int m = blockIdx.z;
  int mBase = blockIdx.x * 64 + wv * 16;
  int nBase = blockIdx.y * 64;
  const u16* A = xb + (size_t)mBase * E_DIM;
  const u16* B = wb + (size_t)m * Nw * E_DIM + (size_t)nBase * E_DIM;

  f32x8 zero = {0.f,0.f,0.f,0.f,0.f,0.f,0.f,0.f};
  f32x8 acc[4] = {zero, zero, zero, zero};
  gemm4_lds(A, E_DIM, B, E_DIM, E_DIM, tid, sB, acc);

  int half = lane >> 4, lr = lane & 15;
  bool odd = (lr & 1) != 0;
#pragma unroll
  for (int n = 0; n < 4; ++n) {
#pragma unroll
    for (int r = 0; r < 8; ++r) {
      int t = mBase + r + 8 * half;
      int f = nBase + n * 16 + lr;
      int j = (f & (HD - 1)) >> 1;
      float c = cosT[t * (HD / 2) + j];
      float s = sinT[t * (HD / 2) + j];
      float v = acc[n][r];
      float p = __shfl_xor(v, 1, 32);           // rope partner (adjacent lane)
      float o = odd ? (p * s + v * c) : (v * c - p * s);
      o *= oscale;
      int head = f >> 5;
      outb[(((size_t)(m * nHeads + head) * T_LEN + t) << 5) + (f & (HD - 1))] = f2bf(o);
    }
  }
}

// ---------------------------------------------------------------------------
// V projection, stored transposed: v_bT[f][t], f in [0,768) = kvh*96 + fin
// ---------------------------------------------------------------------------
__global__ __launch_bounds__(128)
void k_proj_v(const u16* __restrict__ xb, const u16* __restrict__ wb,
              u16* __restrict__ vT) {
  __shared__ u16 sB[2 * 64 * 32];
  int tid = threadIdx.x;
  int lane = tid & 31, wv = tid >> 5;
  int mBase = blockIdx.x * 64 + wv * 16;
  int nBase = blockIdx.y * 64;
  f32x8 zero = {0.f,0.f,0.f,0.f,0.f,0.f,0.f,0.f};
  f32x8 acc[4] = {zero, zero, zero, zero};
  gemm4_lds(xb + (size_t)mBase * E_DIM, E_DIM, wb + (size_t)nBase * E_DIM, E_DIM,
            E_DIM, tid, sB, acc);
  int half = lane >> 4, lr = lane & 15;
#pragma unroll
  for (int n = 0; n < 4; ++n)
#pragma unroll
    for (int r = 0; r < 8; ++r) {
      int t = mBase + r + 8 * half;
      int f = nBase + n * 16 + lr;
      vT[(size_t)f * T_LEN + t] = f2bf(acc[n][r]);
    }
}

// ---------------------------------------------------------------------------
// 3-ensemble flash attention (causal) + linear combine + RMS-norm + subln.
// One wave per (16-query tile, head). Score tile = one WMMA (K=hd=32).
// ---------------------------------------------------------------------------
__global__ __launch_bounds__(32)
void k_attn(const u16* __restrict__ qb, const u16* __restrict__ kb,
            const u16* __restrict__ vT, const float* __restrict__ raw,
            const float* __restrict__ wscale, const float* __restrict__ subln,
            u16* __restrict__ attnb) {
  __shared__ u16 sP[16 * 32];   // P tile staging: C-layout -> A-layout
  int lane = threadIdx.x;
  int half = lane >> 4, lr = lane & 15;
  int qt = blockIdx.x, h = blockIdx.y, kvh = h >> 1;
  int tBase = qt << 4;
  float ws = wscale[0];

  f32x8 zero = {0.f,0.f,0.f,0.f,0.f,0.f,0.f,0.f};
  f32x8 comb[6] = {zero, zero, zero, zero, zero, zero};
  float sln[6];
#pragma unroll
  for (int n = 0; n < 6; ++n) sln[n] = subln[n * 16 + lr];

  int nBlk = (tBase + 16 + 31) >> 5;   // 32-key causal blocks

  for (int m = 0; m < M_ENS; ++m) {
    const u16* qbase = qb + ((size_t)(m * NHEAD + h) * T_LEN + tBase) * HD;
    bf16x16 aq = load_a(qbase, HD, 0, lane);
    const u16* kbase = kb + (size_t)(m * KVH + kvh) * T_LEN * HD;
    const u16* vbase = vT + (size_t)kvh * 96 * T_LEN;

    f32x8 acc[6] = {zero, zero, zero, zero, zero, zero};
    float rmax[8], rsum[8];
#pragma unroll
    for (int r = 0; r < 8; ++r) { rmax[r] = -1e30f; rsum[r] = 0.f; }

    for (int sb = 0; sb < nBlk; ++sb) {
      int s0 = sb * 32;
      bf16x16 bk0 = load_b(kbase + (size_t)s0 * HD, HD, 0, lane);
      bf16x16 bk1 = load_b(kbase + (size_t)(s0 + 16) * HD, HD, 0, lane);
      f32x8 S0 = mma(aq, bk0, zero);
      f32x8 S1 = mma(aq, bk1, zero);

#pragma unroll
      for (int r = 0; r < 8; ++r) {
        int t = tBase + r + 8 * half;
        int key0 = s0 + lr;
        float s0v = (key0      <= t) ? S0[r] : -1e30f;   // causal mask
        float s1v = (key0 + 16 <= t) ? S1[r] : -1e30f;
        float bm = fmaxf(s0v, s1v);
        bm = fmaxf(bm, __shfl_xor(bm, 1, 32));
        bm = fmaxf(bm, __shfl_xor(bm, 2, 32));
        bm = fmaxf(bm, __shfl_xor(bm, 4, 32));
        bm = fmaxf(bm, __shfl_xor(bm, 8, 32));
        float nm = fmaxf(rmax[r], bm);
        float corr = __expf(rmax[r] - nm);
        rmax[r] = nm;
        float p0 = __expf(s0v - nm);
        float p1 = __expf(s1v - nm);
        float ps = p0 + p1;
        ps += __shfl_xor(ps, 1, 32);
        ps += __shfl_xor(ps, 2, 32);
        ps += __shfl_xor(ps, 4, 32);
        ps += __shfl_xor(ps, 8, 32);
        rsum[r] = rsum[r] * corr + ps;
#pragma unroll
        for (int n = 0; n < 6; ++n) acc[n][r] *= corr;
        sP[(r + 8 * half) * 32 + lr]      = f2bf(p0);
        sP[(r + 8 * half) * 32 + 16 + lr] = f2bf(p1);
      }
      __syncthreads();
      // reload P as WMMA A operand (16x32) from LDS
      const u16* pp = &sP[lr * 32 + half * 16];
      BfPack up;
      up.q[0] = *(const int4*)pp;
      up.q[1] = *(const int4*)(pp + 8);
      bf16x16 ap = up.v;
#pragma unroll
      for (int n = 0; n < 6; ++n) {
        bf16x16 bv = load_b(vbase + (size_t)n * 16 * T_LEN, T_LEN, s0, lane);
        acc[n] = mma(ap, bv, acc[n]);
      }
      __syncthreads();
    }

    float wm = tanhf(raw[m]) * ws;   // map weight (linearity over softmax)
#pragma unroll
    for (int r = 0; r < 8; ++r) {
      float inv = wm / rsum[r];
#pragma unroll
      for (int n = 0; n < 6; ++n) comb[n][r] += acc[n][r] * inv;
    }
  }

  // RMS over 96 features + subln scale, store bf16 [t][h*96 + f]
#pragma unroll
  for (int r = 0; r < 8; ++r) {
    float ss = 0.f;
#pragma unroll
    for (int n = 0; n < 6; ++n) ss += comb[n][r] * comb[n][r];
    ss += __shfl_xor(ss, 1, 32);
    ss += __shfl_xor(ss, 2, 32);
    ss += __shfl_xor(ss, 4, 32);
    ss += __shfl_xor(ss, 8, 32);
    float rs = rsqrtf(ss * (1.0f / 96.0f) + 1e-5f);
    int t = tBase + r + 8 * half;
#pragma unroll
    for (int n = 0; n < 6; ++n)
      attnb[(size_t)t * E_DIM + h * 96 + n * 16 + lr] = f2bf(comb[n][r] * rs * sln[n]);
  }
}

// ---------------------------------------------------------------------------
// Output projection: attn(2048x1536) @ out_w^T(1536x1536) -> f32 d_out
// ---------------------------------------------------------------------------
__global__ __launch_bounds__(128)
void k_outproj(const u16* __restrict__ ab, const u16* __restrict__ wb,
               float* __restrict__ out) {
  __shared__ u16 sB[2 * 64 * 32];
  int tid = threadIdx.x;
  int lane = tid & 31, wv = tid >> 5;
  int mBase = blockIdx.x * 64 + wv * 16;
  int nBase = blockIdx.y * 64;
  f32x8 zero = {0.f,0.f,0.f,0.f,0.f,0.f,0.f,0.f};
  f32x8 acc[4] = {zero, zero, zero, zero};
  gemm4_lds(ab + (size_t)mBase * E_DIM, E_DIM, wb + (size_t)nBase * E_DIM, E_DIM,
            E_DIM, tid, sB, acc);
  int half = lane >> 4, lr = lane & 15;
#pragma unroll
  for (int n = 0; n < 4; ++n)
#pragma unroll
    for (int r = 0; r < 8; ++r) {
      int t = mBase + r + 8 * half;
      int f = nBase + n * 16 + lr;
      out[(size_t)t * E_DIM + f] = acc[n][r];
    }
}

// ---------------------------------------------------------------------------
// Host launch
// ---------------------------------------------------------------------------
extern "C" void kernel_launch(void* const* d_in, const int* in_sizes, int n_in,
                              void* d_out, int out_size, void* d_ws, size_t ws_size,
                              hipStream_t stream) {
  (void)in_sizes; (void)n_in; (void)out_size; (void)ws_size;
  const float* x      = (const float*)d_in[0];
  const float* cosT   = (const float*)d_in[1];
  const float* sinT   = (const float*)d_in[2];
  const float* q_w    = (const float*)d_in[3];
  const float* k_w    = (const float*)d_in[4];
  const float* v_w    = (const float*)d_in[5];
  const float* out_w  = (const float*)d_in[6];
  const float* raw    = (const float*)d_in[7];
  const float* wscale = (const float*)d_in[8];
  const float* subln  = (const float*)d_in[9];
  float* out = (float*)d_out;

  // workspace layout (bf16 elements)
  const size_t XB = (size_t)T_LEN * E_DIM;
  const size_t QW = (size_t)M_ENS * 512 * E_DIM;
  const size_t KW = (size_t)M_ENS * 256 * E_DIM;
  const size_t VW = (size_t)768 * E_DIM;
  const size_t OW = (size_t)E_DIM * E_DIM;
  const size_t QB = (size_t)M_ENS * NHEAD * T_LEN * HD;
  const size_t KB = (size_t)M_ENS * KVH * T_LEN * HD;
  const size_t VT = (size_t)KVH * 96 * T_LEN;

  u16* w = (u16*)d_ws;
  u16* xb  = w;            w += XB;
  u16* qwb = w;            w += QW;
  u16* kwb = w;            w += KW;
  u16* vwb = w;            w += VW;
  u16* owb = w;            w += OW;
  u16* qb  = w;            w += QB;
  u16* kb  = w;            w += KB;
  u16* vT  = w;            w += VT;
  u16* ab  = w;            // T_LEN*E_DIM

  // 1) convert f32 -> bf16
  k_cvt<<<(int)((XB + 255) / 256), 256, 0, stream>>>(x, xb, (int)XB);
  k_cvt<<<(int)((QW + 255) / 256), 256, 0, stream>>>(q_w, qwb, (int)QW);
  k_cvt<<<(int)((KW + 255) / 256), 256, 0, stream>>>(k_w, kwb, (int)KW);
  k_cvt<<<(int)((VW + 255) / 256), 256, 0, stream>>>(v_w, vwb, (int)VW);
  k_cvt<<<(int)((OW + 255) / 256), 256, 0, stream>>>(out_w, owb, (int)OW);

  const float qscale = 0.17677669529663687f;  // hd^-0.5, hd=32

  // 2) Q projection + RoPE + scale   (N=512 per ensemble, 16 heads)
  k_proj_rope<<<dim3(T_LEN / 64, 512 / 64, M_ENS), 128, 0, stream>>>(
      xb, qwb, cosT, sinT, qb, 512, NHEAD, qscale);
  // 3) K projection + RoPE           (N=256 per ensemble, 8 kv heads)
  k_proj_rope<<<dim3(T_LEN / 64, 256 / 64, M_ENS), 128, 0, stream>>>(
      xb, kwb, cosT, sinT, kb, 256, KVH, 1.0f);
  // 4) V projection (transposed store)
  k_proj_v<<<dim3(T_LEN / 64, 768 / 64, 1), 128, 0, stream>>>(xb, vwb, vT);

  // 5) 3-ensemble flash attention + combine + RMS norm
  k_attn<<<dim3(T_LEN / 16, NHEAD), 32, 0, stream>>>(qb, kb, vT, raw, wscale,
                                                     subln, ab);

  // 6) output projection -> f32
  k_outproj<<<dim3(T_LEN / 64, E_DIM / 64, 1), 128, 0, stream>>>(ab, owb, out);
}
